// MSTGCNHead_86655260164729
// MI455X (gfx1250) — matrix operand order repaired and checked by reading, not compiled, not run on hardware
//
#include <hip/hip_runtime.h>

// MSTGCN head for MI455X (gfx1250), wave32, WMMA f32_16x16x32_f16.
// Identity adjacency => Chebyshev conv collapses to GEMM with Weff = th0-th1+th2.
// All GEMMs run on v_wmma_f32_16x16x32_f16 (f16 in, f32 accumulate).

typedef __attribute__((ext_vector_type(16))) _Float16 v16h;
typedef __attribute__((ext_vector_type(8)))  _Float16 v8h;
typedef __attribute__((ext_vector_type(4)))  _Float16 v4h;
typedef __attribute__((ext_vector_type(8)))  float    v8f;

union V16H { v16h v; v8h h[2]; };

#define Bdim 32
#define Ndim 512
#define Tdim 12
#define FIN  128
#define CF   64
#define TF   64
#define PRED 12

// A fragment: 16x32 f16 tile, per-lane row pointer rowp (k-contiguous), ka = (lane>>4)*8
__device__ __forceinline__ v16h load_a(const _Float16* rowp, int ka) {
  V16H a;
  a.h[0] = *(const v8h*)(rowp + ka);
  a.h[1] = *(const v8h*)(rowp + ka + 16);
  return a.v;
}
// B fragment: 32x16 f16 tile from K-contiguous column panel, kb = (lane>>4)*16; reads k = kb/2..+7 and kb/2+16..+23 (same order as A)
__device__ __forceinline__ v16h load_b(const _Float16* colp, int kb) {
  V16H b;
  b.h[0] = *(const v8h*)(colp + kb / 2);
  b.h[1] = *(const v8h*)(colp + kb / 2 + 16);
  return b.v;
}

#define WMMA(acc, a, bfrag) \
  acc = __builtin_amdgcn_wmma_f32_16x16x32_f16(false, a, false, bfrag, (short)0, acc, false, false)

// ---------------- f32 -> f16 bulk convert (x), 4-wide ----------------
__global__ void __launch_bounds__(256)
cvt_f32_f16_x4(const float* __restrict__ in, _Float16* __restrict__ out, long n4) {
  long i = (long)blockIdx.x * blockDim.x + threadIdx.x;
  long stride = (long)gridDim.x * blockDim.x;
  for (; i < n4; i += stride) {
    float4 v = ((const float4*)in)[i];
    v4h o;
    o[0] = (_Float16)v.x; o[1] = (_Float16)v.y;
    o[2] = (_Float16)v.z; o[3] = (_Float16)v.w;
    ((v4h*)out)[i] = o;
  }
}

// ---------------- weight panel prep (tiny) ----------------
__global__ void __launch_bounds__(256)
prep_weights(const float* __restrict__ th0, const float* __restrict__ tw0,
             const float* __restrict__ rw0, const float* __restrict__ th1,
             const float* __restrict__ tw1, const float* __restrict__ rw1,
             const float* __restrict__ fw,
             _Float16* __restrict__ W0t, _Float16* __restrict__ R0t,
             _Float16* __restrict__ T0t, _Float16* __restrict__ W1t,
             _Float16* __restrict__ R1t, _Float16* __restrict__ T1t,
             _Float16* __restrict__ Ftw) {
  int tid = threadIdx.x;
  // W0t[o*128+f] = th0[0,f,o] - th0[1,f,o] + th0[2,f,o]
  for (int i = tid; i < 64 * 128; i += 256) {
    int o = i >> 7, f = i & 127;
    float v = th0[f * 64 + o] - th0[8192 + f * 64 + o] + th0[16384 + f * 64 + o];
    W0t[i] = (_Float16)v;
  }
  for (int i = tid; i < 64 * 128; i += 256) R0t[i] = (_Float16)rw0[i]; // (64,128) already o-major
  // T0t[o*192 + dt*64 + c] = tw0[o,c,0,dt]
  for (int i = tid; i < 64 * 192; i += 256) {
    int o = i / 192, r = i % 192, dt = r >> 6, c = r & 63;
    T0t[i] = (_Float16)tw0[o * 192 + c * 3 + dt];
  }
  for (int i = tid; i < 64 * 64; i += 256) {
    int o = i >> 6, f = i & 63;
    float v = th1[f * 64 + o] - th1[4096 + f * 64 + o] + th1[8192 + f * 64 + o];
    W1t[i] = (_Float16)v;
  }
  for (int i = tid; i < 64 * 64; i += 256) R1t[i] = (_Float16)rw1[i];
  for (int i = tid; i < 64 * 192; i += 256) {
    int o = i / 192, r = i % 192, dt = r >> 6, c = r & 63;
    T1t[i] = (_Float16)tw1[o * 192 + c * 3 + dt];
  }
  // Ftw: 16x768, rows 12..15 zero; fw flat (12,12,1,64) already p-major (t,f)-contiguous
  for (int i = tid; i < 16 * 768; i += 256) {
    int p = i / 768;
    Ftw[i] = (p < PRED) ? (_Float16)fw[i] : (_Float16)0.0f;
  }
}

// ---------------- pass S: s = relu(A @ Wt), store (B,T,N,64) f16 ----------------
__global__ void __launch_bounds__(128)
cheb_gemm_kernel(const _Float16* __restrict__ A, const _Float16* __restrict__ Wt,
                 _Float16* __restrict__ S, int Fin,
                 long strideB, long strideT, long strideN) {
  const int ntile = blockIdx.x, t = blockIdx.y, b = blockIdx.z;
  const int lane = threadIdx.x & 31, wave = threadIdx.x >> 5;
  const int otile = wave * 16;
  const int m = lane & 15;
  const int ka = (lane >> 4) * 8, kb = (lane >> 4) * 16, rhalf = (lane >> 4) * 8;
  const int node = ntile * 16 + m;
  const _Float16* arow = A + (long)b * strideB + (long)t * strideT + (long)node * strideN;
  const _Float16* bcol = Wt + (size_t)(otile + m) * Fin;
  v8f acc = (v8f)0.0f;
  for (int k0 = 0; k0 < Fin; k0 += 32) {
    v16h a = load_a(arow + k0, ka);
    v16h w = load_b(bcol + k0, kb);
    WMMA(acc, a, w);
  }
  _Float16* sbase = S + (((size_t)b * Tdim + t) * Ndim) * 64;
  const int col = otile + m;
#pragma unroll
  for (int r = 0; r < 8; ++r) {
    float v = acc[r];
    v = v > 0.0f ? v : 0.0f;
    sbase[(size_t)(ntile * 16 + r + rhalf) * 64 + col] = (_Float16)v;
  }
}

// ---------------- pass H: rconv GEMM + temporal-conv GEMM + bias + relu + LN ----------------
__global__ void __launch_bounds__(128)
block_tail_kernel(const _Float16* __restrict__ A, const _Float16* __restrict__ Rt, int Fin,
                  long strideB, long strideT, long strideN,
                  const _Float16* __restrict__ S, const _Float16* __restrict__ Tt,
                  const float* __restrict__ tb, const float* __restrict__ rb,
                  const float* __restrict__ lg, const float* __restrict__ lb,
                  _Float16* __restrict__ H) {
  __shared__ float lds[16 * 64];
  __shared__ float s_mu[16], s_rs[16];
  const int ntile = blockIdx.x, t = blockIdx.y, b = blockIdx.z;
  const int lane = threadIdx.x & 31, wave = threadIdx.x >> 5;
  const int otile = wave * 16;
  const int m = lane & 15;
  const int ka = (lane >> 4) * 8, kb = (lane >> 4) * 16, rhalf = (lane >> 4) * 8;
  const int node = ntile * 16 + m;
  v8f acc = (v8f)0.0f;
  // residual 1x1 conv: K = Fin over input features
  {
    const _Float16* arow = A + (long)b * strideB + (long)t * strideT + (long)node * strideN;
    const _Float16* bcol = Rt + (size_t)(otile + m) * Fin;
    for (int k0 = 0; k0 < Fin; k0 += 32) {
      v16h a = load_a(arow + k0, ka);
      v16h w = load_b(bcol + k0, kb);
      WMMA(acc, a, w);
    }
  }
  // temporal conv: K = 3*64 over (dt, c); zero-pad at time boundaries
  {
    const _Float16* bcol = Tt + (size_t)(otile + m) * 192;
#pragma unroll
    for (int idx = 0; idx < 6; ++idx) {
      const int dt = idx >> 1;
      const int cbase = (idx & 1) * 32;
      const int ts = t + dt - 1;
      v16h a;
      if (ts >= 0 && ts < Tdim) {
        const _Float16* arow = S + (((size_t)b * Tdim + ts) * Ndim + node) * 64 + cbase;
        a = load_a(arow, ka);
      } else {
        a = (v16h)(_Float16)0.0f;
      }
      v16h w = load_b(bcol + idx * 32, kb);
      WMMA(acc, a, w);
    }
  }
  // bias + relu -> LDS (f32)
  {
    const int oc = otile + m;
    const float bias = tb[oc] + rb[oc];
#pragma unroll
    for (int r = 0; r < 8; ++r) {
      float v = acc[r] + bias;
      v = v > 0.0f ? v : 0.0f;
      lds[(r + rhalf) * 64 + oc] = v;
    }
  }
  __syncthreads();
  if (threadIdx.x < 16) {
    const int nl = threadIdx.x;
    float s = 0.0f, ss = 0.0f;
    for (int o = 0; o < 64; ++o) {
      float v = lds[nl * 64 + o];
      s += v; ss += v * v;
    }
    float mu = s * (1.0f / 64.0f);
    float var = ss * (1.0f / 64.0f) - mu * mu;
    s_mu[nl] = mu;
    s_rs[nl] = rsqrtf(var + 1e-5f);
  }
  __syncthreads();
  {
    const int nl = threadIdx.x >> 3;
    const int ob = (threadIdx.x & 7) * 8;
    const float mu = s_mu[nl], rs = s_rs[nl];
    v8h outv;
#pragma unroll
    for (int j = 0; j < 8; ++j) {
      const int o = ob + j;
      float v = (lds[nl * 64 + o] - mu) * rs * lg[o] + lb[o];
      outv[j] = (_Float16)v;
    }
    *(v8h*)(H + (((size_t)b * Tdim + t) * Ndim + ntile * 16 + nl) * 64 + ob) = outv;
  }
}

// ---------------- head: out[b,n,p] = sum_{t,f} h2[b,t,n,f]*Ftw[p, t*64+f] + fb[p] ----------------
__global__ void __launch_bounds__(32)
head_kernel(const _Float16* __restrict__ H2, const _Float16* __restrict__ Ftw,
            const float* __restrict__ fb, float* __restrict__ out) {
  const int ntile = blockIdx.x, b = blockIdx.y;
  const int lane = threadIdx.x & 31;
  const int m = lane & 15;
  const int ka = (lane >> 4) * 8, kb = (lane >> 4) * 16, rhalf = (lane >> 4) * 8;
  const int node = ntile * 16 + m;
  const _Float16* bcol = Ftw + (size_t)m * 768;
  v8f acc = (v8f)0.0f;
#pragma unroll 4
  for (int idx = 0; idx < 24; ++idx) {
    const int t = idx >> 1;
    const int fbase = (idx & 1) * 32;
    const _Float16* arow = H2 + (((size_t)b * Tdim + t) * Ndim + node) * 64 + fbase;
    v16h a = load_a(arow, ka);
    v16h w = load_b(bcol + idx * 32, kb);
    WMMA(acc, a, w);
  }
  const int p = m;
  if (p < PRED) {
    const float bias = fb[p];
#pragma unroll
    for (int r = 0; r < 8; ++r) {
      const int nn = ntile * 16 + r + rhalf;
      out[((size_t)b * Ndim + nn) * PRED + p] = acc[r] + bias;
    }
  }
}

extern "C" void kernel_launch(void* const* d_in, const int* in_sizes, int n_in,
                              void* d_out, int out_size, void* d_ws, size_t ws_size,
                              hipStream_t stream) {
  const float* x   = (const float*)d_in[0];
  const float* th0 = (const float*)d_in[1];
  const float* tw0 = (const float*)d_in[2];
  const float* tb0 = (const float*)d_in[3];
  const float* rw0 = (const float*)d_in[4];
  const float* rb0 = (const float*)d_in[5];
  const float* lg0 = (const float*)d_in[6];
  const float* lb0 = (const float*)d_in[7];
  const float* th1 = (const float*)d_in[8];
  const float* tw1 = (const float*)d_in[9];
  const float* tb1 = (const float*)d_in[10];
  const float* rw1 = (const float*)d_in[11];
  const float* rb1 = (const float*)d_in[12];
  const float* lg1 = (const float*)d_in[13];
  const float* lb1 = (const float*)d_in[14];
  const float* fw  = (const float*)d_in[15];
  const float* fb  = (const float*)d_in[16];

  char* ws = (char*)d_ws;
  const long XH_ELEMS = (long)Bdim * Ndim * Tdim * FIN;      // 25,165,824
  const long SH_ELEMS = (long)Bdim * Tdim * Ndim * 64;       // 12,582,912
  _Float16* xh = (_Float16*)(ws);
  _Float16* sh = (_Float16*)(ws + XH_ELEMS * 2);
  _Float16* h1 = (_Float16*)(ws + XH_ELEMS * 2 + SH_ELEMS * 2);
  _Float16* h2 = (_Float16*)(ws + XH_ELEMS * 2 + SH_ELEMS * 4);
  _Float16* wb = (_Float16*)(ws + XH_ELEMS * 2 + SH_ELEMS * 6);
  _Float16* W0t = wb;               // 64*128
  _Float16* R0t = W0t + 64 * 128;   // 64*128
  _Float16* T0t = R0t + 64 * 128;   // 64*192
  _Float16* W1t = T0t + 64 * 192;   // 64*64
  _Float16* R1t = W1t + 64 * 64;    // 64*64
  _Float16* T1t = R1t + 64 * 64;    // 64*192
  _Float16* Ftw = T1t + 64 * 192;   // 16*768

  cvt_f32_f16_x4<<<4096, 256, 0, stream>>>(x, xh, XH_ELEMS / 4);
  prep_weights<<<1, 256, 0, stream>>>(th0, tw0, rw0, th1, tw1, rw1, fw,
                                      W0t, R0t, T0t, W1t, R1t, T1t, Ftw);

  dim3 grid(Ndim / 16, Tdim, Bdim);
  // block 1: A = xh (B,N,T,128)
  const long sB0 = (long)Ndim * Tdim * FIN, sT0 = FIN, sN0 = (long)Tdim * FIN;
  cheb_gemm_kernel<<<grid, 128, 0, stream>>>(xh, W0t, sh, FIN, sB0, sT0, sN0);
  block_tail_kernel<<<grid, 128, 0, stream>>>(xh, R0t, FIN, sB0, sT0, sN0,
                                              sh, T0t, tb0, rb0, lg0, lb0, h1);
  // block 2: A = h1 (B,T,N,64)
  const long sB1 = (long)Tdim * Ndim * 64, sT1 = (long)Ndim * 64, sN1 = 64;
  cheb_gemm_kernel<<<grid, 128, 0, stream>>>(h1, W1t, sh, TF, sB1, sT1, sN1);
  block_tail_kernel<<<grid, 128, 0, stream>>>(h1, R1t, TF, sB1, sT1, sN1,
                                              sh, T1t, tb1, rb1, lg1, lb1, h2);
  // head
  head_kernel<<<dim3(Ndim / 16, Bdim), 32, 0, stream>>>(h2, Ftw, fb, (float*)d_out);
}
